// NePuEncoder_28286654612090
// MI455X (gfx1250) — compile-verified, matrix operand order checked by reading, not run on hardware
//
#include <hip/hip_runtime.h>

typedef _Float16 half_t;
typedef __attribute__((ext_vector_type(16))) _Float16 v16h;
typedef __attribute__((ext_vector_type(8)))  _Float16 v8h;
typedef __attribute__((ext_vector_type(8)))  float    v8f;

#define DEVI __device__ __forceinline__

constexpr int kB = 2, kN = 384, kD = 128, kLat = 256;
constexpr int kRows = kB * kN;           // 768
constexpr int kNPos = 33, kKPos = 64;    // pos features, padded K for WMMA
constexpr int kSplit = 4;                // neighbor-axis split-K for occupancy
constexpr int kNPerSplit = kN / kSplit;  // 96 neighbors (6 tiles) per split
constexpr float kEps = 1e-5f;

// ---------------------------------------------------------------------------
// WMMA helpers (CDNA5: V_WMMA_F32_16X16X32_F16, wave32)
// A fragment: lane L holds row M=L%16; 16 halves map to K = kb..kb+7 and
// kb+16..kb+23 with kb = ks + (L/16)*8  -> two contiguous 16B loads.
// B fragment: same packing with N = L%16, sourced from W stored [n][k].
// C/D: v8f, row M = r + 8*(L/16), col N = L%16.
// ---------------------------------------------------------------------------
DEVI v16h frag16(const half_t* __restrict__ row, int kb) {
  const v8h a = *(const v8h*)(row + kb);
  const v8h b = *(const v8h*)(row + kb + 16);
  v16h r;
#pragma unroll
  for (int i = 0; i < 8; ++i) { r[i] = a[i]; r[i + 8] = b[i]; }
  return r;
}

DEVI v8f wmma16(v16h a, v16h b, v8f c) {
  return __builtin_amdgcn_wmma_f32_16x16x32_f16(false, a, false, b, (short)0, c,
                                                false, false);
}

DEVI void zero_acc(v8f (&acc)[2][8]) {
#pragma unroll
  for (int i = 0; i < 2; ++i)
#pragma unroll
    for (int c = 0; c < 8; ++c)
#pragma unroll
      for (int r = 0; r < 8; ++r) acc[i][c][r] = 0.f;
}

// ---------------------------------------------------------------------------
// Generic GEMM: C[M x Nout] = act(A_h[M x K] @ W_h[Nout x K]^T + bias)
// one wave per 16x16 output tile; fragments straight from global (L2-resident)
// ---------------------------------------------------------------------------
__global__ __launch_bounds__(256) void gemm_rt(
    const half_t* __restrict__ A, const half_t* __restrict__ W,
    const float* __restrict__ bias, float* __restrict__ C,
    half_t* __restrict__ Ch, int M, int K, int Nout, int doRelu) {
  const int gw   = (blockIdx.x * blockDim.x + threadIdx.x) >> 5;
  const int lane = threadIdx.x & 31;
  const int lm = lane & 15, lh = lane >> 4;
  const int mtiles = M >> 4;
  const int mt = gw % mtiles;
  const int ct = gw / mtiles;
  v8f acc;
#pragma unroll
  for (int r = 0; r < 8; ++r) acc[r] = 0.f;
  const half_t* arow = A + (size_t)(mt * 16 + lm) * K;
  const half_t* wrow = W + (size_t)(ct * 16 + lm) * K;
  for (int ks = 0; ks < K; ks += 32) {
    const int kb = ks + lh * 8;
    acc = wmma16(frag16(arow, kb), frag16(wrow, kb), acc);
  }
  const int col = ct * 16 + lm;
  const float bv = bias ? bias[col] : 0.f;
#pragma unroll
  for (int r = 0; r < 8; ++r) {
    const int row = mt * 16 + r + lh * 8;
    float v = acc[r] + bv;
    if (doRelu) v = fmaxf(v, 0.f);
    C[(size_t)row * Nout + col] = v;
    if (Ch) Ch[(size_t)row * Nout + col] = (half_t)v;
  }
}

// ---------------------------------------------------------------------------
// Fused vector-attention block, split-K over the neighbor axis.
// blockIdx.x -> (batch, 16-query tile); blockIdx.y -> neighbor split (96 nbrs).
// Per 16-neighbor tile: pos-encode GEMM (K=64) -> edges -> GEMM Wg1 (relu)
// -> GEMM Wg2 -> per-channel online softmax partials (max, denom, numer).
// Partials per (row, split, d) are exactly merged by attn_combine_k.
// All LDS rows are wave-local (wave w owns pair rows 32w..32w+31).
// ---------------------------------------------------------------------------
__global__ __launch_bounds__(256) void attn_block(
    const float* __restrict__ xyz, const float* __restrict__ q,
    const float* __restrict__ k, const float* __restrict__ v,
    const half_t* __restrict__ Wg1h, const half_t* __restrict__ Wg2h,
    const half_t* __restrict__ Wpeh, const float* __restrict__ bpe,
    const float* __restrict__ bg1, const float* __restrict__ bg2,
    float* __restrict__ pmax, float* __restrict__ psum,
    float* __restrict__ pnum) {
  __shared__ half_t sPF[256 * kKPos];   // pos features (A of GEMM1), 32 KB
  __shared__ half_t sPos[256 * kD];     // pos (kept for v+pos), 64 KB
  __shared__ half_t sEH[256 * kD];      // edges, then h, 64 KB

  const int b  = blockIdx.x / (kN / 16);
  const int m0 = (blockIdx.x % (kN / 16)) * 16;
  const int sp = blockIdx.y;
  const int nBeg = sp * kNPerSplit, nEnd = nBeg + kNPerSplit;
  const int tid = threadIdx.x;
  const int lane = tid & 31, w = tid >> 5;
  const int lm = lane & 15, lh = lane >> 4;
  const int rowbase = b * kN;

  // per-lane invariants: q values and biases for this wave's 2 query rows
  float qreg[2][8], bpeR[8], bg1R[8], bg2R[8];
#pragma unroll
  for (int ct = 0; ct < 8; ++ct) {
    const int d = ct * 16 + lm;
    bpeR[ct] = bpe[d]; bg1R[ct] = bg1[d]; bg2R[ct] = bg2[d];
    qreg[0][ct] = q[(size_t)(rowbase + m0 + 2 * w + 0) * kD + d];
    qreg[1][ct] = q[(size_t)(rowbase + m0 + 2 * w + 1) * kD + d];
  }
  // this thread's pair (p = tid): query row plm, neighbor slot pln
  const int plm = tid >> 4, pln = tid & 15;
  const float xq0 = xyz[(size_t)(rowbase + m0 + plm) * 3 + 0] * 4.f;
  const float xq1 = xyz[(size_t)(rowbase + m0 + plm) * 3 + 1] * 4.f;
  const float xq2 = xyz[(size_t)(rowbase + m0 + plm) * 3 + 2] * 4.f;

  float rmax[2][8], rsum[2][8], rnum[2][8];
#pragma unroll
  for (int i = 0; i < 2; ++i)
#pragma unroll
    for (int ct = 0; ct < 8; ++ct) {
      rmax[i][ct] = -3.0e38f; rsum[i][ct] = 0.f; rnum[i][ct] = 0.f;
    }

  const float frq[5] = {1.f, 8.75f, 16.5f, 24.25f, 32.f};

  for (int n0 = nBeg; n0 < nEnd; n0 += 16) {
    // prefetch next tile's k/v rows into near cache (global_prefetch_b8 path)
    if (n0 + 16 < nEnd) {
      const size_t pr = (size_t)(rowbase + n0 + 16 + pln) * kD + plm * 8;
      __builtin_prefetch(&k[pr], 0, 1);
      __builtin_prefetch(&v[pr], 0, 1);
    }
    // ---- positional features for pair p=tid: [pd, sin(pd*f), cos(pd*f)...]
    {
      const float xn0 = xyz[(size_t)(rowbase + n0 + pln) * 3 + 0] * 4.f;
      const float xn1 = xyz[(size_t)(rowbase + n0 + pln) * 3 + 1] * 4.f;
      const float xn2 = xyz[(size_t)(rowbase + n0 + pln) * 3 + 2] * 4.f;
      const float pd0 = xq0 - xn0, pd1 = xq1 - xn1, pd2 = xq2 - xn2;
      half_t* dst = &sPF[tid * kKPos];
      dst[0] = (half_t)pd0; dst[1] = (half_t)pd1; dst[2] = (half_t)pd2;
      int idx = 3;
#pragma unroll
      for (int fi = 0; fi < 5; ++fi) {
        dst[idx + 0] = (half_t)__sinf(pd0 * frq[fi]);
        dst[idx + 1] = (half_t)__sinf(pd1 * frq[fi]);
        dst[idx + 2] = (half_t)__sinf(pd2 * frq[fi]);
        dst[idx + 3] = (half_t)__cosf(pd0 * frq[fi]);
        dst[idx + 4] = (half_t)__cosf(pd1 * frq[fi]);
        dst[idx + 5] = (half_t)__cosf(pd2 * frq[fi]);
        idx += 6;
      }
#pragma unroll
      for (int z = kNPos; z < kKPos; ++z) dst[z] = (half_t)0.f;
    }
    __syncthreads();

    v8f acc[2][8];
    // ---- GEMM1: pos = PF @ Wpe^T   (K = 64)
    zero_acc(acc);
#pragma unroll
    for (int ks = 0; ks < kKPos; ks += 32) {
      const int kb = ks + lh * 8;
      const v16h a0 = frag16(&sPF[(2 * w + 0) * 16 * kKPos + lm * kKPos], kb);
      const v16h a1 = frag16(&sPF[(2 * w + 1) * 16 * kKPos + lm * kKPos], kb);
#pragma unroll
      for (int ct = 0; ct < 8; ++ct) {
        const v16h bf = frag16(Wpeh + (size_t)(ct * 16 + lm) * kKPos, kb);
        acc[0][ct] = wmma16(a0, bf, acc[0][ct]);
        acc[1][ct] = wmma16(a1, bf, acc[1][ct]);
      }
    }
    // epilogue: pos += bpe; edges = q - k + pos  (both stored f16)
#pragma unroll
    for (int i = 0; i < 2; ++i)
#pragma unroll
      for (int ct = 0; ct < 8; ++ct) {
        const int d = ct * 16 + lm;
#pragma unroll
        for (int r = 0; r < 8; ++r) {
          const int ln = r + lh * 8;
          const int p = (2 * w + i) * 16 + ln;
          const float posb = acc[i][ct][r] + bpeR[ct];
          sPos[p * kD + d] = (half_t)posb;
          const float kv = k[(size_t)(rowbase + n0 + ln) * kD + d];
          sEH[p * kD + d] = (half_t)(qreg[i][ct] - kv + posb);
        }
      }
    __syncthreads();

    // ---- GEMM2: h = relu(E @ Wg1^T + bg1)
    zero_acc(acc);
#pragma unroll
    for (int ks = 0; ks < kD; ks += 32) {
      const int kb = ks + lh * 8;
      const v16h a0 = frag16(&sEH[(2 * w + 0) * 16 * kD + lm * kD], kb);
      const v16h a1 = frag16(&sEH[(2 * w + 1) * 16 * kD + lm * kD], kb);
#pragma unroll
      for (int ct = 0; ct < 8; ++ct) {
        const v16h bf = frag16(Wg1h + (size_t)(ct * 16 + lm) * kD, kb);
        acc[0][ct] = wmma16(a0, bf, acc[0][ct]);
        acc[1][ct] = wmma16(a1, bf, acc[1][ct]);
      }
    }
#pragma unroll
    for (int i = 0; i < 2; ++i)
#pragma unroll
      for (int ct = 0; ct < 8; ++ct) {
        const int d = ct * 16 + lm;
#pragma unroll
        for (int r = 0; r < 8; ++r) {
          const int p = (2 * w + i) * 16 + r + lh * 8;
          sEH[p * kD + d] = (half_t)fmaxf(acc[i][ct][r] + bg1R[ct], 0.f);
        }
      }
    __syncthreads();

    // ---- GEMM3: logits = H @ Wg2^T + bg2
    zero_acc(acc);
#pragma unroll
    for (int ks = 0; ks < kD; ks += 32) {
      const int kb = ks + lh * 8;
      const v16h a0 = frag16(&sEH[(2 * w + 0) * 16 * kD + lm * kD], kb);
      const v16h a1 = frag16(&sEH[(2 * w + 1) * 16 * kD + lm * kD], kb);
#pragma unroll
      for (int ct = 0; ct < 8; ++ct) {
        const v16h bf = frag16(Wg2h + (size_t)(ct * 16 + lm) * kD, kb);
        acc[0][ct] = wmma16(a0, bf, acc[0][ct]);
        acc[1][ct] = wmma16(a1, bf, acc[1][ct]);
      }
    }
    // ---- per-channel online softmax over the 16 neighbors in this tile.
    // C-tile rows == neighbors: reduce 8 regs + one shfl_xor(16).
#pragma unroll
    for (int i = 0; i < 2; ++i)
#pragma unroll
      for (int ct = 0; ct < 8; ++ct) {
        const int d = ct * 16 + lm;
        float tmax = -3.0e38f;
#pragma unroll
        for (int r = 0; r < 8; ++r) tmax = fmaxf(tmax, acc[i][ct][r] + bg2R[ct]);
        tmax = fmaxf(tmax, __shfl_xor(tmax, 16, 32));
        const float nm = fmaxf(rmax[i][ct], tmax);
        const float scale = __expf(rmax[i][ct] - nm);
        float s = 0.f, num = 0.f;
#pragma unroll
        for (int r = 0; r < 8; ++r) {
          const int ln = r + lh * 8;
          const int p = (2 * w + i) * 16 + ln;
          const float e = __expf(acc[i][ct][r] + bg2R[ct] - nm);
          const float val = v[(size_t)(rowbase + n0 + ln) * kD + d] +
                            (float)sPos[p * kD + d];
          s += e; num += e * val;
        }
        rsum[i][ct] = rsum[i][ct] * scale + s;
        rnum[i][ct] = rnum[i][ct] * scale + num;
        rmax[i][ct] = nm;
      }
    __syncthreads();
  }

  // write split partials (half-lane partials combined; rmax identical per half)
#pragma unroll
  for (int i = 0; i < 2; ++i)
#pragma unroll
    for (int ct = 0; ct < 8; ++ct) {
      const float S  = rsum[i][ct] + __shfl_xor(rsum[i][ct], 16, 32);
      const float NU = rnum[i][ct] + __shfl_xor(rnum[i][ct], 16, 32);
      if (lh == 0) {
        const int row = rowbase + m0 + 2 * w + i;
        const int d = ct * 16 + lm;
        const size_t o = ((size_t)row * kSplit + sp) * kD + d;
        pmax[o] = rmax[i][ct]; psum[o] = S; pnum[o] = NU;
      }
    }
}

// exact merge of split-K softmax partials + residual add -> pre-BN y
__global__ void attn_combine_k(const float* __restrict__ pmax,
                               const float* __restrict__ psum,
                               const float* __restrict__ pnum,
                               const float* __restrict__ f,
                               float* __restrict__ y) {
  const int i = blockIdx.x * blockDim.x + threadIdx.x;
  if (i >= kRows * kD) return;
  const int row = i >> 7, d = i & 127;
  const size_t base = (size_t)row * kSplit * kD + d;
  float gm = -3.0e38f;
#pragma unroll
  for (int s = 0; s < kSplit; ++s) gm = fmaxf(gm, pmax[base + (size_t)s * kD]);
  float S = 0.f, NU = 0.f;
#pragma unroll
  for (int s = 0; s < kSplit; ++s) {
    const size_t o = base + (size_t)s * kD;
    const float sc = __expf(pmax[o] - gm);
    S += psum[o] * sc;
    NU += pnum[o] * sc;
  }
  y[i] = NU / S + f[i];
}

// ---------------------------------------------------------------------------
// Small elementwise / reduction kernels
// ---------------------------------------------------------------------------
__global__ void encoder_k(const float* __restrict__ feats,
                          const float* __restrict__ encW,
                          const float* __restrict__ encb,
                          float* __restrict__ f, half_t* __restrict__ fh) {
  const int i = blockIdx.x * blockDim.x + threadIdx.x;
  if (i >= kRows * kD) return;
  const int row = i >> 7, d = i & 127;
  const float v = feats[row] * encW[d] + encb[d];
  f[i] = v; fh[i] = (half_t)v;
}

__global__ void tohalf_k(const float* __restrict__ s, half_t* __restrict__ d,
                         int n) {
  const int i = blockIdx.x * blockDim.x + threadIdx.x;
  if (i < n) d[i] = (half_t)s[i];
}

__global__ void pad_wpe_k(const float* __restrict__ s, half_t* __restrict__ d) {
  const int i = blockIdx.x * blockDim.x + threadIdx.x;
  if (i >= 3 * kD * kKPos) return;
  const int kk = i & (kKPos - 1);
  const int rowAll = i >> 6;  // block*128 + dout
  d[i] = (kk < kNPos) ? (half_t)s[rowAll * kNPos + kk] : (half_t)0.f;
}

// BatchNorm (train-mode batch stats) two-stage reduction, deterministic
__global__ void bn_part_k(const float* __restrict__ x,
                          const float* __restrict__ add,
                          float* __restrict__ part) {
  const int d = threadIdx.x;        // 128 threads
  const int bi = blockIdx.x;        // 48 blocks, 16 rows each
  float s = 0.f, sq = 0.f;
#pragma unroll
  for (int r = 0; r < 16; ++r) {
    const size_t idx = (size_t)(bi * 16 + r) * kD + d;
    float v = x[idx];
    if (add) v += add[idx];
    s += v; sq += v * v;
  }
  part[bi * 2 * kD + d] = s;
  part[bi * 2 * kD + kD + d] = sq;
}

__global__ void bn_final_k(const float* __restrict__ part,
                           float* __restrict__ mv) {
  const int d = threadIdx.x;  // 128 threads
  float s = 0.f, sq = 0.f;
#pragma unroll
  for (int bi = 0; bi < kRows / 16; ++bi) {
    s += part[bi * 2 * kD + d];
    sq += part[bi * 2 * kD + kD + d];
  }
  const float m = s / kRows;
  mv[d] = m;
  mv[kD + d] = sq / kRows - m * m;
}

__global__ void bn_apply_k(const float* __restrict__ x,
                           const float* __restrict__ add,
                           const float* __restrict__ mv,
                           const float* __restrict__ g,
                           const float* __restrict__ bb,
                           float* __restrict__ f, half_t* __restrict__ fh) {
  const int i = blockIdx.x * blockDim.x + threadIdx.x;
  if (i >= kRows * kD) return;
  const int d = i & 127;
  float v = x[i];
  if (add) v += add[i];
  v = (v - mv[d]) * rsqrtf(mv[kD + d] + kEps) * g[d] + bb[d];
  f[i] = v; fh[i] = (half_t)v;
}

__global__ void maxpool_k(const float* __restrict__ e, float* __restrict__ out) {
  const int i = blockIdx.x * blockDim.x + threadIdx.x;
  if (i >= kB * kLat) return;
  const int b = i / kLat, c = i % kLat;
  float m = -3.0e38f;
  for (int n = 0; n < kN; ++n)
    m = fmaxf(m, e[(size_t)(b * kN + n) * kLat + c]);
  out[i] = m;
}

// ---------------------------------------------------------------------------
// Host orchestration
// ---------------------------------------------------------------------------
extern "C" void kernel_launch(void* const* d_in, const int* in_sizes, int n_in,
                              void* d_out, int out_size, void* d_ws,
                              size_t ws_size, hipStream_t stream) {
  (void)in_sizes; (void)n_in; (void)out_size; (void)ws_size;
  const float* xyz   = (const float*)d_in[0];
  const float* feats = (const float*)d_in[1];
  const float* encW  = (const float*)d_in[2];
  const float* encb  = (const float*)d_in[3];
  const float* tbWq  = (const float*)d_in[4];
  const float* tbWk  = (const float*)d_in[5];
  const float* tbWv  = (const float*)d_in[6];
  const float* tbWg1 = (const float*)d_in[7];
  const float* tbbg1 = (const float*)d_in[8];
  const float* tbWg2 = (const float*)d_in[9];
  const float* tbbg2 = (const float*)d_in[10];
  const float* tbWpe = (const float*)d_in[11];
  const float* tbbpe = (const float*)d_in[12];
  const float* tbg   = (const float*)d_in[13];
  const float* tbbt  = (const float*)d_in[14];
  const float* emW1  = (const float*)d_in[15];
  const float* emb1  = (const float*)d_in[16];
  const float* emW2  = (const float*)d_in[17];
  const float* emb2  = (const float*)d_in[18];
  const float* emg   = (const float*)d_in[19];
  const float* embt  = (const float*)d_in[20];
  const float* fW1   = (const float*)d_in[21];
  const float* fb1   = (const float*)d_in[22];
  const float* fW2   = (const float*)d_in[23];
  const float* fb2   = (const float*)d_in[24];

  char* p = (char*)d_ws;
  auto alloc = [&](size_t bytes) -> char* {
    char* r = p;
    p += (bytes + 255) & ~(size_t)255;
    return r;
  };
  const size_t fsz = (size_t)kRows * kD;
  const size_t psz = (size_t)kRows * kSplit * kD;
  float*  f    = (float*)alloc(fsz * 4);
  half_t* fh   = (half_t*)alloc(fsz * 2);
  float*  q    = (float*)alloc(fsz * 4);
  float*  kk   = (float*)alloc(fsz * 4);
  float*  vv   = (float*)alloc(fsz * 4);
  float*  y    = (float*)alloc(fsz * 4);
  float*  y1   = (float*)alloc(fsz * 4);
  half_t* y1h  = (half_t*)alloc(fsz * 2);
  float*  y2   = (float*)alloc(fsz * 4);
  float*  e1   = (float*)alloc((size_t)kRows * kLat * 4);
  half_t* e1h  = (half_t*)alloc((size_t)kRows * kLat * 2);
  float*  e2   = (float*)alloc((size_t)kRows * kLat * 4);
  float*  mv   = (float*)alloc(2 * kD * 4);
  float*  part = (float*)alloc((size_t)(kRows / 16) * 2 * kD * 4);
  float*  pmax = (float*)alloc(psz * 4);
  float*  psum = (float*)alloc(psz * 4);
  float*  pnum = (float*)alloc(psz * 4);
  half_t* Wqh  = (half_t*)alloc((size_t)3 * kD * kD * 2);
  half_t* Wkh  = (half_t*)alloc((size_t)3 * kD * kD * 2);
  half_t* Wvh  = (half_t*)alloc((size_t)3 * kD * kD * 2);
  half_t* Wg1h = (half_t*)alloc((size_t)3 * kD * kD * 2);
  half_t* Wg2h = (half_t*)alloc((size_t)3 * kD * kD * 2);
  half_t* Wpeh = (half_t*)alloc((size_t)3 * kD * kKPos * 2);
  half_t* eW1h = (half_t*)alloc((size_t)2 * kD * kD * 2);
  half_t* eW2h = (half_t*)alloc((size_t)2 * kD * kD * 2);
  half_t* fW1h = (half_t*)alloc((size_t)kLat * kD * 2);
  half_t* fW2h = (half_t*)alloc((size_t)kLat * kLat * 2);

  const int TH = 256;
  const int n3 = 3 * kD * kD;
  tohalf_k<<<(n3 + TH - 1) / TH, TH, 0, stream>>>(tbWq, Wqh, n3);
  tohalf_k<<<(n3 + TH - 1) / TH, TH, 0, stream>>>(tbWk, Wkh, n3);
  tohalf_k<<<(n3 + TH - 1) / TH, TH, 0, stream>>>(tbWv, Wvh, n3);
  tohalf_k<<<(n3 + TH - 1) / TH, TH, 0, stream>>>(tbWg1, Wg1h, n3);
  tohalf_k<<<(n3 + TH - 1) / TH, TH, 0, stream>>>(tbWg2, Wg2h, n3);
  pad_wpe_k<<<(3 * kD * kKPos + TH - 1) / TH, TH, 0, stream>>>(tbWpe, Wpeh);
  const int n2 = 2 * kD * kD;
  tohalf_k<<<(n2 + TH - 1) / TH, TH, 0, stream>>>(emW1, eW1h, n2);
  tohalf_k<<<(n2 + TH - 1) / TH, TH, 0, stream>>>(emW2, eW2h, n2);
  tohalf_k<<<(kLat * kD + TH - 1) / TH, TH, 0, stream>>>(fW1, fW1h, kLat * kD);
  tohalf_k<<<(kLat * kLat + TH - 1) / TH, TH, 0, stream>>>(fW2, fW2h, kLat * kLat);

  encoder_k<<<(int)((fsz + TH - 1) / TH), TH, 0, stream>>>(feats, encW, encb, f, fh);

  const int gElem = (int)((fsz + TH - 1) / TH);  // 384
  // tiles for [768 x 128]: (768/16)*(128/16)=384 tiles -> 48 blocks of 8 waves
  for (int i = 0; i < 3; ++i) {
    const size_t wo = (size_t)i * kD * kD;
    gemm_rt<<<48, 256, 0, stream>>>(fh, Wqh + wo, nullptr, q, nullptr, kRows, kD, kD, 0);
    gemm_rt<<<48, 256, 0, stream>>>(fh, Wkh + wo, nullptr, kk, nullptr, kRows, kD, kD, 0);
    gemm_rt<<<48, 256, 0, stream>>>(fh, Wvh + wo, nullptr, vv, nullptr, kRows, kD, kD, 0);
    attn_block<<<dim3(kB * (kN / 16), kSplit), 256, 0, stream>>>(
        xyz, q, kk, vv, Wg1h + wo, Wg2h + wo, Wpeh + (size_t)i * kD * kKPos,
        tbbpe + i * kD, tbbg1 + i * kD, tbbg2 + i * kD, pmax, psum, pnum);
    attn_combine_k<<<gElem, TH, 0, stream>>>(pmax, psum, pnum, f, y);
    bn_part_k<<<kRows / 16, kD, 0, stream>>>(y, nullptr, part);
    bn_final_k<<<1, kD, 0, stream>>>(part, mv);
    bn_apply_k<<<gElem, TH, 0, stream>>>(
        y, nullptr, mv, tbg + i * kD, tbbt + i * kD, f, fh);
    if (i > 0) {
      const int j = i - 1;
      const size_t eo = (size_t)j * kD * kD;
      gemm_rt<<<48, 256, 0, stream>>>(fh, eW1h + eo, emb1 + j * kD, y1, y1h, kRows, kD, kD, 1);
      gemm_rt<<<48, 256, 0, stream>>>(y1h, eW2h + eo, emb2 + j * kD, y2, nullptr, kRows, kD, kD, 1);
      bn_part_k<<<kRows / 16, kD, 0, stream>>>(f, y2, part);
      bn_final_k<<<1, kD, 0, stream>>>(part, mv);
      bn_apply_k<<<gElem, TH, 0, stream>>>(
          f, y2, mv, emg + j * kD, embt + j * kD, f, fh);
    }
  }

  // final head: [768x128]@[128->256] relu, then [768x256]@[256->256], max over N
  gemm_rt<<<96, 256, 0, stream>>>(fh, fW1h, fb1, e1, e1h, kRows, kD, kLat, 1);
  gemm_rt<<<96, 256, 0, stream>>>(e1h, fW2h, fb2, e2, nullptr, kRows, kLat, kLat, 0);
  maxpool_k<<<(kB * kLat + TH - 1) / TH, TH, 0, stream>>>(e2, (float*)d_out);
}
